// unit_gcn_81655918232209
// MI455X (gfx1250) — matrix operand order repaired and checked by reading, not compiled, not run on hardware
//
#include <hip/hip_runtime.h>

#define N_   64
#define CIN  64
#define COUT 64
#define T_   128
#define V_   25
#define VP   32      // padded V for fragment-friendly layouts
#define S_   3
#define R_   8
#define BN_EPS 1e-5f

typedef __attribute__((ext_vector_type(16))) __bf16 v16bf;
typedef __attribute__((ext_vector_type(8)))  float  v8f;

union BF16Frag { v16bf v; unsigned short u[16]; };

// native bf16 convert (lowers to v_cvt_pk_bf16_f32 on gfx1250)
__device__ __forceinline__ unsigned short f2bf(float f) {
    __bf16 h = (__bf16)f;
    return __builtin_bit_cast(unsigned short, h);
}

// ---------------- Kernel 1: temporal mean  xm[n,c,v] = mean_t x[n,c,t,v] ----
__global__ void k_mean(const float* __restrict__ x, float* __restrict__ xm) {
    int i = blockIdx.x * blockDim.x + threadIdx.x;
    if (i >= N_ * CIN * V_) return;
    int v  = i % V_;
    int nc = i / V_;
    const float* p = x + (size_t)nc * T_ * V_ + v;
    float s = 0.f;
#pragma unroll 8
    for (int t = 0; t < T_; ++t) s += p[t * V_];
    xm[i] = s * (1.0f / T_);
}

// ---------------- Kernel 2: build padded M (bf16, 32x32 tiles) + ybias -----
// one workgroup per (n,s).  Mb layout: [n][s][o][u:32][v:32] bf16, pads = 0.
__global__ void k_buildM(const float* __restrict__ xm,
                         const float* __restrict__ W1s, const float* __restrict__ b1s,
                         const float* __restrict__ W2s, const float* __restrict__ b2s,
                         const float* __restrict__ W4s, const float* __restrict__ b4s,
                         const float* __restrict__ PA,  const float* __restrict__ alpha_p,
                         const float* __restrict__ b3s,
                         unsigned short* __restrict__ Mb, float* __restrict__ ybias) {
    __shared__ float sx1[R_ * V_];
    __shared__ float sx2[R_ * V_];
    __shared__ float sdiff[R_ * V_ * V_];
    int n = blockIdx.x / S_;
    int s = blockIdx.x % S_;
    int tid = threadIdx.x;
    float alpha = alpha_p[0];

    if (tid < R_ * V_) {
        int r = tid / V_, v = tid % V_;
        float a1 = b1s[s * R_ + r];
        float a2 = b2s[s * R_ + r];
        for (int c = 0; c < CIN; ++c) {
            float xv = xm[(n * CIN + c) * V_ + v];
            a1 += xv * W1s[(s * R_ + r) * CIN + c];
            a2 += xv * W2s[(s * R_ + r) * CIN + c];
        }
        sx1[tid] = a1;
        sx2[tid] = a2;
    }
    __syncthreads();
    for (int i = tid; i < R_ * V_ * V_; i += blockDim.x) {
        int r = i / (V_ * V_);
        int rem = i % (V_ * V_);
        int u = rem / V_, v = rem % V_;
        sdiff[i] = tanhf(sx1[r * V_ + u] - sx2[r * V_ + v]);
    }
    __syncthreads();
    // one padded (o,u) row of M per task (u in [0,32))
    for (int task = tid; task < COUT * VP; task += blockDim.x) {
        int o = task / VP, u = task % VP;
        unsigned short* mrowp =
            Mb + ((((size_t)n * S_ + s) * COUT + o) * VP + u) * VP;
        if (u < V_) {
            float b4 = b4s[s * COUT + o];
            float w4[R_];
#pragma unroll
            for (int r = 0; r < R_; ++r) w4[r] = W4s[(s * COUT + o) * R_ + r];
            float rowsum = 0.f;
            for (int v = 0; v < V_; ++v) {
                float acc = 0.f;
#pragma unroll
                for (int r = 0; r < R_; ++r)
                    acc += w4[r] * sdiff[(r * V_ + u) * V_ + v];
                float m = (acc + b4) * alpha + PA[(s * V_ + u) * V_ + v];
                rowsum += m;
                mrowp[v] = f2bf(m);
            }
#pragma unroll
            for (int v = V_; v < VP; ++v) mrowp[v] = 0;   // zero pad cols
            atomicAdd(&ybias[(n * COUT + o) * V_ + u], b3s[s * COUT + o] * rowsum);
        } else {
#pragma unroll
            for (int v = 0; v < VP; ++v) mrowp[v] = 0;    // zero pad rows
        }
    }
}

// ---------------- Kernel 3: fused WMMA GEMMs + BN stats --------------------
// grid (T_/16, N_), 256 threads = 8 waves.
// LDS: W3 bf16 (192x64) + x3 bf16 tile (192 x 16t x 32v, pad zeroed).
__global__ void k_gcn_wmma(const float* __restrict__ x,
                           const float* __restrict__ W3s,
                           const unsigned short* __restrict__ Mb,
                           const float* __restrict__ ybias,
                           float* __restrict__ y,
                           float* __restrict__ gSum, float* __restrict__ gSqs) {
    extern __shared__ unsigned char smem[];
    unsigned short* sW3 = (unsigned short*)smem;          // 192*64 bf16
    unsigned short* sX3 = sW3 + (S_ * COUT * CIN);        // 192*16*32 bf16
    int n    = blockIdx.y;
    int tb   = blockIdx.x;            // 16-wide t block
    int tid  = threadIdx.x;
    int lane = tid & 31;
    int wave = tid >> 5;              // 0..7
    int mrow  = lane & 15;            // A-matrix row within tile
    int kbase = (lane < 16) ? 0 : 8;  // unused; A/B loads use k = 16*(lane/16) + e
    int ncol  = lane & 15;            // B/C/D column within tile

    // stage W3 (S*COUT x CIN) into LDS as bf16
    for (int i = tid; i < S_ * COUT * CIN; i += blockDim.x)
        sW3[i] = f2bf(W3s[i]);
    // zero the v-pad region of sX3 (cols 25..31 of each (so,t) row)
    for (int i = tid; i < S_ * COUT * 16; i += blockDim.x) {
        unsigned short* row = sX3 + (size_t)i * VP;
#pragma unroll
        for (int p = V_; p < VP; ++p) row[p] = 0;
    }
    __syncthreads();

    // ---- Stage 1: x3[so, t, v] = sum_c W3[so,c] * x[n,c,t,v]  (K = 64) ----
    for (int j = wave; j < 25; j += 8) {        // 25 column tiles of 16 (t*v = 400)
        BF16Frag B0, B1;
        int q = j * 16 + ncol;                  // flattened (t,v) column
        int t = q / V_, v = q % V_;
        const float* xp = x + ((size_t)(n * CIN) * T_ + tb * 16 + t) * V_ + v;
#pragma unroll
        for (int e = 0; e < 16; ++e) {
            int k = (lane < 16) ? e : 16 + e;   // B-matrix K index
            B0.u[e] = f2bf(xp[(size_t)k * (T_ * V_)]);
            B1.u[e] = f2bf(xp[(size_t)(k + 32) * (T_ * V_)]);
        }
        for (int i = 0; i < 12; ++i) {          // 12 so-row tiles (192/16)
            BF16Frag A0, A1;
#pragma unroll
            for (int e = 0; e < 16; ++e) {
                int k = (lane < 16) ? e : 16 + e;
                A0.u[e] = sW3[(i * 16 + mrow) * CIN + k];
                A1.u[e] = sW3[(i * 16 + mrow) * CIN + k + 32];
            }
            v8f acc = {};
            acc = __builtin_amdgcn_wmma_f32_16x16x32_bf16(false, A0.v, false, B0.v,
                                                          (short)0, acc, false, false);
            acc = __builtin_amdgcn_wmma_f32_16x16x32_bf16(false, A1.v, false, B1.v,
                                                          (short)0, acc, false, false);
#pragma unroll
            for (int r = 0; r < 8; ++r) {
                int so = i * 16 + ((lane < 16) ? r : 8 + r);
                sX3[((so * 16 + t) * VP) + v] = f2bf(acc[r]);
            }
        }
    }
    __syncthreads();

    // ---- Stage 2: y[o,u,t] = sum_s sum_v M[n,s,o,u,v] * x3[s,o,t,v] -------
    // Each wave owns 8 output channels; the 3 per-s x3 B-fragments are loaded
    // once per channel and reused for both u-tiles (B depends only on (s,o)).
    for (int o = wave; o < COUT; o += 8) {
        BF16Frag Bs[S_];
#pragma unroll
        for (int s = 0; s < S_; ++s) {
#pragma unroll
            for (int e = 0; e < 16; ++e) {
                int kv = (lane < 16) ? e : 16 + e;                    // k = v
                Bs[s].u[e] = sX3[((s * COUT + o) * 16 + ncol) * VP + kv];
            }
        }
        const float* ybp = ybias + (n * COUT + o) * V_;
        float* yp = y + (((size_t)n * COUT + o) * T_ + tb * 16 + ncol) * V_;
        float lsum = 0.f, lsq = 0.f;
#pragma unroll
        for (int uT = 0; uT < 2; ++uT) {
            int u0 = uT * 16 + mrow;             // A row for this lane
            v8f acc = {};
#pragma unroll
            for (int s = 0; s < S_; ++s) {
                BF16Frag A;
                const unsigned short* mp =
                    Mb + (((size_t)(n * S_ + s) * COUT + o) * VP) * VP;
#pragma unroll
                for (int e = 0; e < 16; ++e) {
                    int k = (lane < 16) ? e : 16 + e; // k = v
                    A.u[e] = mp[u0 * VP + k];    // padded: no guards
                }
                acc = __builtin_amdgcn_wmma_f32_16x16x32_bf16(false, A.v, false, Bs[s].v,
                                                              (short)0, acc, false, false);
            }
            // add analytic b3 contribution, store y, gather BN stats
#pragma unroll
            for (int r = 0; r < 8; ++r) {
                int u = uT * 16 + ((lane < 16) ? r : 8 + r);
                if (u < V_) {
                    float val = acc[r] + ybp[u];
                    yp[u] = val;
                    lsum += val;
                    lsq  += val * val;
                }
            }
        }
        for (int off = 16; off > 0; off >>= 1) {
            lsum += __shfl_xor(lsum, off, 32);
            lsq  += __shfl_xor(lsq,  off, 32);
        }
        if (lane == 0) {
            atomicAdd(&gSum[o], lsum);
            atomicAdd(&gSqs[o], lsq);
        }
    }
}

// ---------------- Kernel 4: fold BN stats into scale/shift -----------------
__global__ void k_bnparams(const float* __restrict__ gSum, const float* __restrict__ gSqs,
                           const float* __restrict__ gamma, const float* __restrict__ beta,
                           float* __restrict__ scale, float* __restrict__ shift) {
    int o = threadIdx.x;
    if (o < COUT) {
        const float cnt = (float)(N_ * T_ * V_);
        float mu  = gSum[o] / cnt;
        float var = gSqs[o] / cnt - mu * mu;
        float a   = gamma[o] * rsqrtf(var + BN_EPS);
        scale[o] = a;
        shift[o] = beta[o] - mu * a;
    }
}

// ---------------- Kernel 5: BN apply + residual + ReLU ---------------------
__global__ void k_epilogue(const float4* __restrict__ y4, const float4* __restrict__ x4,
                           const float* __restrict__ scale, const float* __restrict__ shift,
                           float4* __restrict__ out4, int total4) {
    int i = blockIdx.x * blockDim.x + threadIdx.x;
    if (i >= total4) return;
    int c = ((i * 4) / (T_ * V_)) % COUT;   // T_*V_ = 3200, divisible by 4
    float a = scale[c], b = shift[c];
    float4 yv = y4[i], xv = x4[i], o;
    o.x = fmaxf(a * yv.x + b + xv.x, 0.f);
    o.y = fmaxf(a * yv.y + b + xv.y, 0.f);
    o.z = fmaxf(a * yv.z + b + xv.z, 0.f);
    o.w = fmaxf(a * yv.w + b + xv.w, 0.f);
    out4[i] = o;
}

// ---------------- launcher -------------------------------------------------
extern "C" void kernel_launch(void* const* d_in, const int* in_sizes, int n_in,
                              void* d_out, int out_size, void* d_ws, size_t ws_size,
                              hipStream_t stream) {
    const float* x     = (const float*)d_in[0];
    const float* W1s   = (const float*)d_in[1];
    const float* b1s   = (const float*)d_in[2];
    const float* W2s   = (const float*)d_in[3];
    const float* b2s   = (const float*)d_in[4];
    const float* W3s   = (const float*)d_in[5];
    const float* b3s   = (const float*)d_in[6];
    const float* W4s   = (const float*)d_in[7];
    const float* b4s   = (const float*)d_in[8];
    const float* PA    = (const float*)d_in[9];
    const float* alpha = (const float*)d_in[10];
    const float* gamma = (const float*)d_in[11];
    const float* beta  = (const float*)d_in[12];

    unsigned char* ws = (unsigned char*)d_ws;
    size_t off = 0;
    auto alloc = [&](size_t bytes) -> void* {
        void* p = ws + off;
        off = (off + bytes + 255) & ~(size_t)255;
        return p;
    };
    float*          xm    = (float*)alloc((size_t)N_ * CIN * V_ * 4);
    unsigned short* Mb    = (unsigned short*)alloc((size_t)N_ * S_ * COUT * VP * VP * 2);
    float*          ybias = (float*)alloc((size_t)N_ * COUT * V_ * 4);
    float*          y     = (float*)alloc((size_t)N_ * COUT * T_ * V_ * 4);
    float*          gSum  = (float*)alloc(COUT * 4);
    float*          gSqs  = (float*)alloc(COUT * 4);
    float*          scale = (float*)alloc(COUT * 4);
    float*          shift = (float*)alloc(COUT * 4);

    hipMemsetAsync(ybias, 0, (size_t)N_ * COUT * V_ * 4, stream);
    hipMemsetAsync(gSum,  0, COUT * 4, stream);
    hipMemsetAsync(gSqs,  0, COUT * 4, stream);

    k_mean<<<(N_ * CIN * V_ + 255) / 256, 256, 0, stream>>>(x, xm);
    k_buildM<<<N_ * S_, 256, 0, stream>>>(xm, W1s, b1s, W2s, b2s, W4s, b4s,
                                          PA, alpha, b3s, Mb, ybias);
    size_t smem = (size_t)(S_ * COUT * CIN + S_ * COUT * 16 * VP) * sizeof(unsigned short);
    k_gcn_wmma<<<dim3(T_ / 16, N_), 256, smem, stream>>>(x, W3s, Mb, ybias, y, gSum, gSqs);
    k_bnparams<<<1, 64, 0, stream>>>(gSum, gSqs, gamma, beta, scale, shift);
    int total4 = N_ * COUT * T_ * V_ / 4;
    k_epilogue<<<(total4 + 255) / 256, 256, 0, stream>>>((const float4*)y, (const float4*)x,
                                                         scale, shift, (float4*)d_out, total4);
}
